// NeuroSAT_51573967290668
// MI455X (gfx1250) — matrix operand order repaired and checked by reading, not compiled, not run on hardware
//
#include <hip/hip_runtime.h>
#include <hip/hip_bf16.h>

// ---------------- problem constants (match reference) ----------------
constexpr int N_P   = 200000;
constexpr int N_C   = 50000;
constexpr int D     = 64;
constexpr int VOCAB = 400;
constexpr int NUM_ITERS = 2;   // reference: inp["num_iters"] = 2

constexpr int KC = 320;        // clause A width: 4*D vars + D hidden
constexpr int KP = 128;        // point  A width: D msg + D hidden
constexpr int C_STRIDE = 328;  // LDS row strides (multiples of 8 bf16 = 16B)
constexpr int P_STRIDE = 136;
constexpr int L_STRIDE = 72;

typedef __bf16 bhalf;
typedef __attribute__((ext_vector_type(16))) __bf16 v16bf;
typedef __attribute__((ext_vector_type(8)))  __bf16 v8bf;
typedef __attribute__((ext_vector_type(4)))  __bf16 v4bf;
typedef __attribute__((ext_vector_type(8)))  float  v8f;

__device__ __forceinline__ bhalf f2bf(float f) {
  unsigned u = __builtin_bit_cast(unsigned, f);
  unsigned r = u + 0x7FFFu + ((u >> 16) & 1u);   // round-to-nearest-even
  unsigned short s = (unsigned short)(r >> 16);
  return __builtin_bit_cast(bhalf, s);
}
__device__ __forceinline__ v4bf f2bf4(float4 f) {
  v4bf r;
  r[0] = f2bf(f.x); r[1] = f2bf(f.y); r[2] = f2bf(f.z); r[3] = f2bf(f.w);
  return r;
}
__device__ __forceinline__ float sigf(float x) { return 1.0f / (1.0f + __expf(-x)); }

// ---------------- one-time weight fragment conversion ----------------
// B-fragment layout for v_wmma_f32_16x16x32_bf16 (32x16 K-by-N tile):
// lane l holds column n = l&15; lanes 0-15 hold K 0..7,16..23, lanes 16-31 K 8..15,24..31,
// frag[(tile)*512 + lane*16 + j] = B[kbase + (j<8 ? (l>>4)*8 + j : 16 + (l>>4)*8 + j-8)][n]

__global__ void k_conv_wc(const float* __restrict__ Wih_c,
                          const float* __restrict__ Whh_c,
                          bhalf* __restrict__ wc) {
  int idx = blockIdx.x * 256 + threadIdx.x;
  if (idx >= 4 * 10 * 16 * 512) return;
  int j    = idx & 15;
  int lane = (idx >> 4) & 31;
  int rest = idx >> 9;
  int ntile = rest & 15; rest >>= 4;
  int ks = rest % 10;
  int t  = rest / 10;
  int n = ntile * 16 + (lane & 15);
  int k = ks * 32 + (j < 8 ? (lane >> 4) * 8 + j : 16 + (lane >> 4) * 8 + (j - 8));
  float v = (k < 256) ? Wih_c[((t * 256) + n) * 256 + k]
                      : Whh_c[((t * 256) + n) * 64 + (k - 256)];
  wc[idx] = f2bf(v);
}

__global__ void k_conv_wu(const float* __restrict__ Wih_u,
                          const float* __restrict__ Whh_u,
                          bhalf* __restrict__ wu) {
  int idx = blockIdx.x * 256 + threadIdx.x;
  if (idx >= 4 * 16 * 512) return;
  int j    = idx & 15;
  int lane = (idx >> 4) & 31;
  int rest = idx >> 9;
  int ntile = rest & 15;
  int ks = rest >> 4;
  int n = ntile * 16 + (lane & 15);
  int k = ks * 32 + (j < 8 ? (lane >> 4) * 8 + j : 16 + (lane >> 4) * 8 + (j - 8));
  float v = (k < 64) ? Wih_u[n * 64 + k] : Whh_u[n * 64 + (k - 64)];
  wu[idx] = f2bf(v);
}

__global__ void k_conv_emb(const float* __restrict__ embedding,
                           bhalf* __restrict__ embf) {
  int idx = blockIdx.x * 256 + threadIdx.x;
  if (idx >= 2 * 25 * 512) return;
  int j    = idx & 15;
  int lane = (idx >> 4) & 31;
  int rest = idx >> 9;
  int nt = rest % 25;
  int ks = rest / 25;
  int n = nt * 16 + (lane & 15);
  int k = ks * 32 + (j < 8 ? (lane >> 4) * 8 + j : 16 + (lane >> 4) * 8 + (j - 8));
  embf[idx] = f2bf(embedding[n * 64 + k]);
}

__global__ void k_bias(const float* __restrict__ bih_c, const float* __restrict__ bhh_c,
                       const float* __restrict__ bih_u, const float* __restrict__ bhh_u,
                       float* __restrict__ bias_c, float* __restrict__ bias_u) {
  int idx = blockIdx.x * 256 + threadIdx.x;
  if (idx < 1024)      bias_c[idx] = bih_c[idx] + bhh_c[idx];
  else if (idx < 1280) bias_u[idx - 1024] = bih_u[idx - 1024] + bhh_u[idx - 1024];
}

// ---------------- state init (float4 vectorized) ----------------
__global__ void k_init_p(const float* __restrict__ embedding,
                         const float* __restrict__ x_p_rand,
                         const int* __restrict__ y, const int* __restrict__ p_t,
                         float* __restrict__ x_p, float* __restrict__ x_p_h) {
  int idx = blockIdx.x * 256 + threadIdx.x;      // one float4 each
  if (idx >= N_P * 16) return;
  int i = idx >> 4, d4 = (idx & 15) * 4;
  float4 v = (p_t[i] != 0) ? *(const float4*)&embedding[y[i] * 64 + d4]
                           : *(const float4*)&x_p_rand[i * 64 + d4];
  *(float4*)&x_p[i * 64 + d4]   = v;
  *(float4*)&x_p_h[i * 64 + d4] = make_float4(0.f, 0.f, 0.f, 0.f);
}

__global__ void k_init_c(const float* __restrict__ W_cinit,
                         const float* __restrict__ b_cinit,
                         float* __restrict__ x_c, float* __restrict__ x_c_h) {
  int idx = blockIdx.x * 256 + threadIdx.x;
  if (idx >= N_C * 16) return;
  int d4 = (idx & 15) * 4;
  float4 w = *(const float4*)&W_cinit[d4];
  float4 b = *(const float4*)&b_cinit[d4];
  *(float4*)&x_c[idx * 4]   = make_float4(w.x + b.x, w.y + b.y, w.z + b.z, w.w + b.w);
  *(float4*)&x_c_h[idx * 4] = make_float4(0.f, 0.f, 0.f, 0.f);
}

// ---------------- clause LSTM (4 types computed, per-row select) ----------------
__global__ __launch_bounds__(128)
void k_clause(const float* __restrict__ x_p, float* __restrict__ x_c,
              float* __restrict__ x_c_h, const int* __restrict__ c_t,
              const int* __restrict__ p2c, const bhalf* __restrict__ wc,
              const float* __restrict__ bias_c) {
  __shared__ __align__(16) bhalf A[16 * C_STRIDE];
  __shared__ float Gt[4 * 16 * 64];   // [gate][row][d]
  __shared__ float bias_s[4 * 256];
  __shared__ int   ct_s[16];
  __shared__ int   idx_s[64];

  const int tid  = threadIdx.x;
  const int blk  = blockIdx.x;
  const int lane = tid & 31;
  const int w    = tid >> 5;          // wave id == gate id (i,f,g,o)

  if (tid < 64) idx_s[tid] = p2c[blk * 64 + tid];
  if (tid < 16) ct_s[tid]  = c_t[blk * 16 + tid];
  for (int e = tid; e < 1024; e += 128) bias_s[e] = bias_c[e];
  __syncthreads();

  // gather A = [x_p[p2c[c,0..3]] (256) | x_c[c] (64)] as bf16, float4 at a time
  for (int e = tid; e < 16 * (KC / 4); e += 128) {
    int r = e / 80, c4 = (e % 80) * 4;
    float4 v;
    if (c4 < 256) v = *(const float4*)&x_p[idx_s[r * 4 + (c4 >> 6)] * 64 + (c4 & 63)];
    else          v = *(const float4*)&x_c[(blk * 16 + r) * 64 + (c4 - 256)];
    *(v4bf*)&A[r * C_STRIDE + c4] = f2bf4(v);
  }
  __syncthreads();

  v8f acc[4][4] = {};                 // [type][ntile within wave]
  const int m   = lane & 15;
  const int hi  = lane >> 4;
  const int sel = hi * 8;

  for (int ks = 0; ks < 10; ++ks) {
    const int kbase = ks * 32;
    v16bf a;
    {
      v8bf lo = *(const v8bf*)&A[m * C_STRIDE + kbase + sel];
      v8bf hj = *(const v8bf*)&A[m * C_STRIDE + kbase + 16 + sel];
      #pragma unroll
      for (int j = 0; j < 8; ++j) { a[j] = lo[j]; a[j + 8] = hj[j]; }
    }
    #pragma unroll
    for (int t = 0; t < 4; ++t) {
      if (t == 3 && (ks == 6 || ks == 7)) continue;  // midpoint zeroes slot 3
      #pragma unroll
      for (int n = 0; n < 4; ++n) {
        const int ntile = w * 4 + n;
        const v16bf b = *(const v16bf*)&wc[((((t * 10 + ks) * 16) + ntile) * 32 + lane) * 16];
        acc[t][n] = __builtin_amdgcn_wmma_f32_16x16x32_bf16(
            false, a, false, b, (short)0, acc[t][n], false, false);
      }
    }
  }

  // per-row clause-type select + bias -> LDS gates
  #pragma unroll
  for (int n = 0; n < 4; ++n) {
    const int dloc = n * 16 + m;
    #pragma unroll
    for (int v = 0; v < 8; ++v) {
      const int row = v + hi * 8;
      const int ct  = ct_s[row];
      float g = (ct == 0) ? acc[0][n][v]
              : (ct == 1) ? acc[1][n][v]
              : (ct == 2) ? acc[2][n][v]
              :             acc[3][n][v];
      g += bias_s[ct * 256 + w * 64 + dloc];
      Gt[(w * 16 + row) * 64 + dloc] = g;
    }
  }
  __syncthreads();

  // LSTM combine, in-place state update (block owns its 16 rows)
  for (int e = tid; e < 16 * 16; e += 128) {
    int row = e >> 4, d0 = (e & 15) * 4;
    int base = (blk * 16 + row) * 64 + d0;
    float4 cold = *(const float4*)&x_c_h[base];
    float4 cn4, hn4;
    float* cnp = &cn4.x; float* hnp = &hn4.x; const float* cop = &cold.x;
    #pragma unroll
    for (int q = 0; q < 4; ++q) {
      int d = d0 + q;
      float ig = Gt[(0 * 16 + row) * 64 + d];
      float fg = Gt[(1 * 16 + row) * 64 + d];
      float gg = Gt[(2 * 16 + row) * 64 + d];
      float og = Gt[(3 * 16 + row) * 64 + d];
      float cn = sigf(fg) * cop[q] + sigf(ig) * tanhf(gg);
      cnp[q] = cn;
      hnp[q] = sigf(og) * tanhf(cn);
    }
    *(float4*)&x_c[base]   = hn4;
    *(float4*)&x_c_h[base] = cn4;
  }
}

// ---------------- scatter: msg = adj.T @ x_c ----------------
__global__ void k_zero(float* __restrict__ msg) {
  int idx = blockIdx.x * 256 + threadIdx.x;
  if (idx < N_P * 16)
    *(float4*)&msg[idx * 4] = make_float4(0.f, 0.f, 0.f, 0.f);
}

__global__ void k_scatter(const float* __restrict__ x_c,
                          const int* __restrict__ p2c,
                          float* __restrict__ msg) {
  int idx = blockIdx.x * 256 + threadIdx.x;
  if (idx >= N_C * 4 * 64) return;
  int c = idx >> 8, j = (idx >> 6) & 3, d = idx & 63;
  atomicAdd(&msg[p2c[c * 4 + j] * 64 + d], x_c[c * 64 + d]);
}

// ---------------- point LSTM ----------------
__global__ __launch_bounds__(128)
void k_point(const float* __restrict__ msg, float* __restrict__ x_p,
             float* __restrict__ x_p_h, const int* __restrict__ p_t,
             const bhalf* __restrict__ wu, const float* __restrict__ bias_u) {
  __shared__ __align__(16) bhalf A[16 * P_STRIDE];
  __shared__ float Gt[4 * 16 * 64];
  __shared__ float bias_s[256];
  __shared__ int   pt_s[16];
  const int tid = threadIdx.x, blk = blockIdx.x;
  const int lane = tid & 31, w = tid >> 5;
  const int p0 = blk * 16;

  if (tid < 16) pt_s[tid] = p_t[p0 + tid];
  for (int e = tid; e < 256; e += 128) bias_s[e] = bias_u[e];

  // gather A = [msg (64) | x_p (64)] as bf16, float4 at a time
  for (int e = tid; e < 16 * (KP / 4); e += 128) {
    int r = e >> 5, c4 = (e & 31) * 4;
    float4 v = (c4 < 64) ? *(const float4*)&msg[(p0 + r) * 64 + c4]
                         : *(const float4*)&x_p[(p0 + r) * 64 + (c4 - 64)];
    *(v4bf*)&A[r * P_STRIDE + c4] = f2bf4(v);
  }
  __syncthreads();

  v8f acc[4] = {};
  const int m = lane & 15, hi = lane >> 4, sel = hi * 8;
  for (int ks = 0; ks < 4; ++ks) {
    const int kbase = ks * 32;
    v16bf a;
    {
      v8bf lo = *(const v8bf*)&A[m * P_STRIDE + kbase + sel];
      v8bf hj = *(const v8bf*)&A[m * P_STRIDE + kbase + 16 + sel];
      #pragma unroll
      for (int j = 0; j < 8; ++j) { a[j] = lo[j]; a[j + 8] = hj[j]; }
    }
    #pragma unroll
    for (int n = 0; n < 4; ++n) {
      const int ntile = w * 4 + n;
      const v16bf b = *(const v16bf*)&wu[(((ks * 16) + ntile) * 32 + lane) * 16];
      acc[n] = __builtin_amdgcn_wmma_f32_16x16x32_bf16(
          false, a, false, b, (short)0, acc[n], false, false);
    }
  }

  #pragma unroll
  for (int n = 0; n < 4; ++n) {
    const int dloc = n * 16 + m;
    #pragma unroll
    for (int v = 0; v < 8; ++v) {
      const int row = v + hi * 8;
      Gt[(w * 16 + row) * 64 + dloc] = acc[n][v] + bias_s[w * 64 + dloc];
    }
  }
  __syncthreads();

  for (int e = tid; e < 16 * 16; e += 128) {
    int row = e >> 4, d0 = (e & 15) * 4;
    if (pt_s[row] != 0) continue;     // fixed points keep state
    int base = (p0 + row) * 64 + d0;
    float4 cold = *(const float4*)&x_p_h[base];
    float4 cn4, hn4;
    float* cnp = &cn4.x; float* hnp = &hn4.x; const float* cop = &cold.x;
    #pragma unroll
    for (int q = 0; q < 4; ++q) {
      int d = d0 + q;
      float ig = Gt[(0 * 16 + row) * 64 + d];
      float fg = Gt[(1 * 16 + row) * 64 + d];
      float gg = Gt[(2 * 16 + row) * 64 + d];
      float og = Gt[(3 * 16 + row) * 64 + d];
      float cn = sigf(fg) * cop[q] + sigf(ig) * tanhf(gg);
      cnp[q] = cn;
      hnp[q] = sigf(og) * tanhf(cn);
    }
    *(float4*)&x_p[base]   = hn4;
    *(float4*)&x_p_h[base] = cn4;
  }
}

// ---------------- tied classifier: logits = x_p @ emb.T + cls_bias ----------------
__global__ __launch_bounds__(128)
void k_logits(const float* __restrict__ x_p, const bhalf* __restrict__ embf,
              const float* __restrict__ cls_bias, float* __restrict__ out) {
  __shared__ __align__(16) bhalf A[16 * L_STRIDE];
  const int tid = threadIdx.x, blk = blockIdx.x;
  const int lane = tid & 31, w = tid >> 5;
  const int p0 = blk * 16;

  for (int e = tid; e < 16 * 16; e += 128) {
    int r = e >> 4, c4 = (e & 15) * 4;
    float4 v = *(const float4*)&x_p[(p0 + r) * 64 + c4];
    *(v4bf*)&A[r * L_STRIDE + c4] = f2bf4(v);
  }
  __syncthreads();

  const int m = lane & 15, hi = lane >> 4, sel = hi * 8;
  v16bf a[2];
  #pragma unroll
  for (int ks = 0; ks < 2; ++ks) {
    v8bf lo = *(const v8bf*)&A[m * L_STRIDE + ks * 32 + sel];
    v8bf hj = *(const v8bf*)&A[m * L_STRIDE + ks * 32 + 16 + sel];
    #pragma unroll
    for (int j = 0; j < 8; ++j) { a[ks][j] = lo[j]; a[ks][j + 8] = hj[j]; }
  }

  for (int nt = w; nt < 25; nt += 4) {
    v8f acc = {};
    #pragma unroll
    for (int ks = 0; ks < 2; ++ks) {
      const v16bf b = *(const v16bf*)&embf[(((ks * 25) + nt) * 32 + lane) * 16];
      acc = __builtin_amdgcn_wmma_f32_16x16x32_bf16(
          false, a[ks], false, b, (short)0, acc, false, false);
    }
    const int ncol = nt * 16 + m;
    const float cb = cls_bias[ncol];
    #pragma unroll
    for (int v = 0; v < 8; ++v) {
      const int row = v + hi * 8;
      out[(size_t)(p0 + row) * VOCAB + ncol] = acc[v] + cb;
    }
  }
}

// ---------------- host orchestration ----------------
extern "C" void kernel_launch(void* const* d_in, const int* in_sizes, int n_in,
                              void* d_out, int out_size, void* d_ws, size_t ws_size,
                              hipStream_t stream) {
  const float* embedding = (const float*)d_in[0];
  const float* cls_bias  = (const float*)d_in[1];
  const float* W_cinit   = (const float*)d_in[2];
  const float* b_cinit   = (const float*)d_in[3];
  const float* Wih_c     = (const float*)d_in[4];
  const float* Whh_c     = (const float*)d_in[5];
  const float* bih_c     = (const float*)d_in[6];
  const float* bhh_c     = (const float*)d_in[7];
  const float* Wih_u     = (const float*)d_in[8];
  const float* Whh_u     = (const float*)d_in[9];
  const float* bih_u     = (const float*)d_in[10];
  const float* bhh_u     = (const float*)d_in[11];
  const float* x_p_rand  = (const float*)d_in[12];
  const int*   y         = (const int*)d_in[13];
  const int*   p_t       = (const int*)d_in[14];
  const int*   c_t       = (const int*)d_in[15];
  const int*   p2c       = (const int*)d_in[16];
  (void)in_sizes; (void)n_in; (void)out_size; (void)ws_size;

  char* ws = (char*)d_ws;
  size_t off = 0;
  auto alloc = [&](size_t bytes) -> void* {
    void* p = ws + off;
    off = (off + bytes + 255) & ~(size_t)255;
    return p;
  };
  float* x_p    = (float*)alloc((size_t)N_P * D * 4);
  float* x_p_h  = (float*)alloc((size_t)N_P * D * 4);
  float* msg    = (float*)alloc((size_t)N_P * D * 4);
  float* x_c    = (float*)alloc((size_t)N_C * D * 4);
  float* x_c_h  = (float*)alloc((size_t)N_C * D * 4);
  bhalf* wc     = (bhalf*)alloc((size_t)4 * 10 * 16 * 512 * 2);
  bhalf* wu     = (bhalf*)alloc((size_t)4 * 16 * 512 * 2);
  bhalf* embf   = (bhalf*)alloc((size_t)2 * 25 * 512 * 2);
  float* bias_c = (float*)alloc(1024 * 4);
  float* bias_u = (float*)alloc(256 * 4);

  // weight / bias prep
  k_conv_wc <<<(4 * 10 * 16 * 512 + 255) / 256, 256, 0, stream>>>(Wih_c, Whh_c, wc);
  k_conv_wu <<<(4 * 16 * 512 + 255) / 256,      256, 0, stream>>>(Wih_u, Whh_u, wu);
  k_conv_emb<<<(2 * 25 * 512 + 255) / 256,      256, 0, stream>>>(embedding, embf);
  k_bias    <<<(1280 + 255) / 256,              256, 0, stream>>>(bih_c, bhh_c, bih_u, bhh_u, bias_c, bias_u);

  // state init
  k_init_p<<<(N_P * 16 + 255) / 256, 256, 0, stream>>>(embedding, x_p_rand, y, p_t, x_p, x_p_h);
  k_init_c<<<(N_C * 16 + 255) / 256, 256, 0, stream>>>(W_cinit, b_cinit, x_c, x_c_h);

  for (int it = 0; it < NUM_ITERS; ++it) {
    k_clause <<<N_C / 16, 128, 0, stream>>>(x_p, x_c, x_c_h, c_t, p2c, wc, bias_c);
    k_zero   <<<(N_P * 16 + 255) / 256, 256, 0, stream>>>(msg);
    k_scatter<<<(N_C * 4 * 64 + 255) / 256, 256, 0, stream>>>(x_c, p2c, msg);
    k_point  <<<N_P / 16, 128, 0, stream>>>(msg, x_p, x_p_h, p_t, wu, bias_u);
  }

  k_logits<<<N_P / 16, 128, 0, stream>>>(x_p, embf, cls_bias, (float*)d_out);
}
